// GINEncoder_60576218742836
// MI455X (gfx1250) — matrix operand order repaired and checked by reading, not compiled, not run on hardware
//
#include <hip/hip_runtime.h>

#define D 128
#define BN_EPS 1e-5f

typedef __attribute__((ext_vector_type(16))) __bf16 v16bf;
typedef __attribute__((ext_vector_type(8)))  float  v8f;

union Frag {
    unsigned int u[8];
    v16bf v;
};

__device__ __forceinline__ unsigned short f2bf(float f) {
    unsigned int u = __float_as_uint(f);
    u += 0x7FFFu + ((u >> 16) & 1u);   // round-to-nearest-even
    return (unsigned short)(u >> 16);
}

// ---------------------------------------------------------------- utilities
__global__ void zero_f32(float* __restrict__ p, int n4) {
    int i = blockIdx.x * blockDim.x + threadIdx.x;
    if (i < n4) ((float4*)p)[i] = make_float4(0.f, 0.f, 0.f, 0.f);
}

// Convert W (f32, [mat][k][n] row-major) -> bf16 [mat][n][k] (pairs along k contiguous)
__global__ void pack_w(const float* __restrict__ W, unsigned short* __restrict__ out, int total) {
    int i = blockIdx.x * blockDim.x + threadIdx.x;
    if (i >= total) return;
    int mat = i >> 14;           // / (128*128)
    int rem = i & 16383;
    int k   = rem >> 7;
    int n   = rem & 127;
    out[(size_t)mat * 16384 + (size_t)n * 128 + k] = f2bf(W[i]);
}

// ---------------------------------------------------------------- scatter: agg[dst] += h[src]
__global__ void scatter_add_edges(const float* __restrict__ h,
                                  const long long* __restrict__ ei,
                                  float* __restrict__ agg, int E) {
    long long gid = (long long)blockIdx.x * blockDim.x + threadIdx.x;
    if (gid >= (long long)E * 32) return;
    int e = (int)(gid >> 5);
    int c = ((int)gid & 31) << 2;
    long long s = ei[e];
    long long d = ei[(long long)E + e];
    const float4 v = *(const float4*)&h[s * D + c];
    float* o = &agg[d * D + c];
    atomicAdd(o + 0, v.x);
    atomicAdd(o + 1, v.y);
    atomicAdd(o + 2, v.z);
    atomicAdd(o + 3, v.w);
}

// ---------------------------------------------------------------- pooled[batch[i]] += h[i]
__global__ void pool_nodes(const float* __restrict__ h,
                           const long long* __restrict__ batch,
                           float* __restrict__ pooled, int Nn) {
    long long gid = (long long)blockIdx.x * blockDim.x + threadIdx.x;
    if (gid >= (long long)Nn * 32) return;
    int node = (int)(gid >> 5);
    int c = ((int)gid & 31) << 2;
    long long b = batch[node];
    const float4 v = *(const float4*)&h[(long long)node * D + c];
    float* o = &pooled[b * D + c];
    atomicAdd(o + 0, v.x);
    atomicAdd(o + 1, v.y);
    atomicAdd(o + 2, v.z);
    atomicAdd(o + 3, v.w);
}

// ---------------------------------------------------------------- fused GEMM + (BN+ReLU | bias)
// Out[M,128] = epilogue( (A (+Askip)) @ W + b ), W given as bf16 [n][k] packed pairs.
// mode 0: BN(eval)+ReLU ; mode 1: bias only.
__global__ __launch_bounds__(256)
void gemm_wmma(const float* __restrict__ A,
               const float* __restrict__ Askip,            // nullable (GIN skip: +h)
               const unsigned short* __restrict__ Wbf,     // [128][128] n-major, bf16
               const float* __restrict__ bias,
               const float* __restrict__ gamma,
               const float* __restrict__ beta,
               const float* __restrict__ bmean,
               const float* __restrict__ bvar,
               float* __restrict__ Out,
               int M, int mode) {
    // 136-ushort row pitch: 272B = 68 dwords -> bank stride 4, 16 rows hit 16 distinct banks
    __shared__ unsigned short ldsA[128 * 136];

    const int tid  = threadIdx.x;
    const int row0 = blockIdx.x * 128;

    // ---- stage A (+skip) tile into LDS as bf16 -------------------------------------------
    for (int idx = tid; idx < 128 * 32; idx += 256) {
        int r = idx >> 5;
        int c = (idx & 31) << 2;
        float4 v = make_float4(0.f, 0.f, 0.f, 0.f);
        int gr = row0 + r;
        if (gr < M) {
            v = *(const float4*)&A[(long long)gr * D + c];
            if (Askip) {
                float4 s = *(const float4*)&Askip[(long long)gr * D + c];
                v.x += s.x; v.y += s.y; v.z += s.z; v.w += s.w;
            }
        }
        int o = r * 136 + c;
        ldsA[o + 0] = f2bf(v.x);
        ldsA[o + 1] = f2bf(v.y);
        ldsA[o + 2] = f2bf(v.z);
        ldsA[o + 3] = f2bf(v.w);
    }
    __syncthreads();

    const int wave = tid >> 5;        // 8 waves, 16 rows each
    const int lane = tid & 31;        // wave32
    const int lhi  = lane >> 4;       // 0 / 1 (lane half)
    const int llo  = lane & 15;

    v8f acc[8];
#pragma unroll
    for (int i = 0; i < 8; ++i)
        acc[i] = (v8f){0.f, 0.f, 0.f, 0.f, 0.f, 0.f, 0.f, 0.f};

    const unsigned int* Wu = (const unsigned int*)Wbf;    // 64 dwords per n-column

#pragma unroll
    for (int ks = 0; ks < 4; ++ks) {
        const int k0 = ks * 32;

        // A fragment (ISA 16-bit A 16x32 layout): row = llo, lane-half picks K {0..7|8..15}, then +16
        Frag a;
        const int arow = wave * 16 + llo;
        const int kb   = k0 + (lhi << 3);
#pragma unroll
        for (int i = 0; i < 4; ++i) {
            a.u[i]     = *(const unsigned int*)&ldsA[arow * 136 + kb + 2 * i];
            a.u[4 + i] = *(const unsigned int*)&ldsA[arow * 136 + kb + 16 + 2 * i];
        }

#pragma unroll
        for (int nt = 0; nt < 8; ++nt) {
            // B fragment: lane llo = column n, lane-half picks K {0..7|8..15}, then +16 (matches A)
            Frag b;
            const int ncol = nt * 16 + llo;
            const int kd   = (k0 >> 1) + (lhi << 2);
#pragma unroll
            for (int i = 0; i < 8; ++i)
                b.u[i] = Wu[ncol * 64 + kd + i + (i >= 4 ? 4 : 0)];

            acc[nt] = __builtin_amdgcn_wmma_f32_16x16x32_bf16(
                false, a.v, false, b.v, (short)0, acc[nt], false, false);
        }
    }

    // ---- epilogue: C/D layout -> VGPR r, lane-half gives M = r + 8*lhi, N = llo -----------
#pragma unroll
    for (int nt = 0; nt < 8; ++nt) {
        const int n = nt * 16 + llo;
        float sc, sh;
        if (mode == 0) {
            sc = gamma[n] * rsqrtf(bvar[n] + BN_EPS);
            sh = beta[n] + (bias[n] - bmean[n]) * sc;
        } else {
            sc = 1.f;
            sh = bias[n];
        }
#pragma unroll
        for (int r = 0; r < 8; ++r) {
            int m = row0 + wave * 16 + (lhi << 3) + r;
            if (m < M) {
                float y = acc[nt][r] * sc + sh;
                if (mode == 0) y = fmaxf(y, 0.f);
                Out[(long long)m * D + n] = y;
            }
        }
    }
}

// ---------------------------------------------------------------- driver
extern "C" void kernel_launch(void* const* d_in, const int* in_sizes, int n_in,
                              void* d_out, int out_size, void* d_ws, size_t ws_size,
                              hipStream_t stream) {
    (void)n_in; (void)ws_size;
    const float*     x     = (const float*)d_in[0];
    const long long* ei    = (const long long*)d_in[1];
    const long long* batch = (const long long*)d_in[2];
    const float* W1  = (const float*)d_in[3];
    const float* b1  = (const float*)d_in[4];
    const float* g1  = (const float*)d_in[5];
    const float* be1 = (const float*)d_in[6];
    const float* m1  = (const float*)d_in[7];
    const float* v1  = (const float*)d_in[8];
    const float* W2  = (const float*)d_in[9];
    const float* b2  = (const float*)d_in[10];
    const float* g2  = (const float*)d_in[11];
    const float* be2 = (const float*)d_in[12];
    const float* m2  = (const float*)d_in[13];
    const float* v2  = (const float*)d_in[14];
    const float* Wl  = (const float*)d_in[15];
    const float* bl  = (const float*)d_in[16];

    const int Nn = in_sizes[0] / D;        // 100000 nodes
    const int E  = in_sizes[1] / 2;        // 1.6M edges
    const int Gn = out_size / D;           // 512 graphs
    const int L  = in_sizes[3] / (D * D);  // 4 layers

    // ---- workspace partition ----
    char* ws = (char*)d_ws;
    size_t off = 0;
    auto alloc = [&](size_t bytes) -> char* {
        char* p = ws + off;
        off = (off + bytes + 255) & ~(size_t)255;
        return p;
    };
    float* agg    = (float*)alloc((size_t)Nn * D * 4);
    float* z1     = (float*)alloc((size_t)Nn * D * 4);
    float* hbuf   = (float*)alloc((size_t)Nn * D * 4);
    float* pooled = (float*)alloc((size_t)Gn * D * 4);
    unsigned short* wbf1 = (unsigned short*)alloc((size_t)L * D * D * 2);
    unsigned short* wbf2 = (unsigned short*)alloc((size_t)L * D * D * 2);
    unsigned short* wbfl = (unsigned short*)alloc((size_t)D * D * 2);

    // ---- weight packing (bf16, column-major pairs) ----
    const int tot1 = L * D * D;
    pack_w<<<(tot1 + 255) / 256, 256, 0, stream>>>(W1, wbf1, tot1);
    pack_w<<<(tot1 + 255) / 256, 256, 0, stream>>>(W2, wbf2, tot1);
    pack_w<<<(D * D + 255) / 256, 256, 0, stream>>>(Wl, wbfl, D * D);

    const float* hcur = x;
    const int gemmBlocks = (Nn + 127) / 128;
    const int eBlocks = (int)(((long long)E * 32 + 255) / 256);
    const int zBlocks = (Nn * D / 4 + 255) / 256;

    for (int l = 0; l < L; ++l) {
        zero_f32<<<zBlocks, 256, 0, stream>>>(agg, Nn * D / 4);
        scatter_add_edges<<<eBlocks, 256, 0, stream>>>(hcur, ei, agg, E);
        gemm_wmma<<<gemmBlocks, 256, 0, stream>>>(
            agg, hcur, wbf1 + (size_t)l * D * D,
            b1 + l * D, g1 + l * D, be1 + l * D, m1 + l * D, v1 + l * D,
            z1, Nn, 0);
        gemm_wmma<<<gemmBlocks, 256, 0, stream>>>(
            z1, nullptr, wbf2 + (size_t)l * D * D,
            b2 + l * D, g2 + l * D, be2 + l * D, m2 + l * D, v2 + l * D,
            hbuf, Nn, 0);
        hcur = hbuf;
    }

    zero_f32<<<(Gn * D / 4 + 255) / 256, 256, 0, stream>>>(pooled, Gn * D / 4);
    pool_nodes<<<(int)(((long long)Nn * 32 + 255) / 256), 256, 0, stream>>>(hcur, batch, pooled, Nn);
    gemm_wmma<<<(Gn + 127) / 128, 256, 0, stream>>>(
        pooled, nullptr, wbfl, bl, bl, bl, bl, bl, (float*)d_out, Gn, 1);
}
